// MegaCRN_7387343749713
// MI455X (gfx1250) — matrix operand order repaired and checked by reading, not compiled, not run on hardware
//
#include <hip/hip_runtime.h>
#include <hip/hip_bf16.h>

typedef __attribute__((ext_vector_type(16))) __bf16 v16bf;
typedef __attribute__((ext_vector_type(8)))  __bf16 v8bf;
typedef __attribute__((ext_vector_type(8)))  float  v8f;

#define NP   512      // padded node count (500 -> 512)
#define NN   500
#define BB   64
#define TT   12
#define HENC 64
#define HDEC 128
#define MEMN 20
#define MEMD 64
#define SMAT ((size_t)NP * NP)   // one cheb matrix, elements

#define WMMA_BF16(a, b, c) \
  __builtin_amdgcn_wmma_f32_16x16x32_bf16(false, (a), false, (b), (short)0, (c), false, false)

// ---------------- fragment loads (CDNA5 ISA 7.12.2 layouts) ----------------
// A fragment (16x32, row-major source): lane r=lane&15 is row; lanes<16 carry
// K {0..7,16..23}, lanes>=16 carry K {8..15,24..31}.
__device__ __forceinline__ v16bf ldA(const __bf16* ap) {
  v16bf a;
  ((v8bf*)&a)[0] = *(const v8bf*)(ap);
  ((v8bf*)&a)[1] = *(const v8bf*)(ap + 16);
  return a;
}
// B fragment (32x16, from N-major/transposed source): lane c=lane&15 is col;
// lanes<16 carry K {0..7,16..23}, lanes>=16 carry K {8..15,24..31}.
__device__ __forceinline__ v16bf ldB(const __bf16* bp) {
  v16bf b;
  ((v8bf*)&b)[0] = *(const v8bf*)(bp);
  ((v8bf*)&b)[1] = *(const v8bf*)(bp + 16);
  return b;
}

// ---------------- WMMA core: one wave computes a 32x64 strip ----------------
// Software-pipelined: chunk k+32 fragments are issued before chunk k's WMMAs,
// so 12 loads stay in flight while the matrix pipe works.
__device__ __forceinline__ void wmma_32x64(
    const __bf16* __restrict__ A, int lda,
    const __bf16* __restrict__ Bt, int ldb,
    int row0, int col0, int K, int lane, v8f* acc /*8*/)
{
  const int r   = lane & 15;
  const int kbA = (lane >> 4) << 3;
  const int kbB = (lane >> 4) << 3;
  const __bf16* ap0 = A + (size_t)(row0 + r) * lda + kbA;
  const __bf16* ap1 = ap0 + (size_t)16 * lda;
  const __bf16* bp  = Bt + (size_t)(col0 + r) * ldb + kbB;
  const size_t bs = (size_t)16 * ldb;

  v16bf a0 = ldA(ap0), a1 = ldA(ap1);
  v16bf b0 = ldB(bp), b1 = ldB(bp + bs), b2 = ldB(bp + 2 * bs), b3 = ldB(bp + 3 * bs);

  for (int k0 = 0; k0 < K; k0 += 32) {
    // prefetch next chunk (tail reloads chunk 0; results unused)
    const int kn = (k0 + 32 < K) ? (k0 + 32) : 0;
    v16bf a0n = ldA(ap0 + kn), a1n = ldA(ap1 + kn);
    v16bf b0n = ldB(bp + kn),          b1n = ldB(bp + bs + kn);
    v16bf b2n = ldB(bp + 2 * bs + kn), b3n = ldB(bp + 3 * bs + kn);

    acc[0] = WMMA_BF16(a0, b0, acc[0]);
    acc[1] = WMMA_BF16(a0, b1, acc[1]);
    acc[2] = WMMA_BF16(a0, b2, acc[2]);
    acc[3] = WMMA_BF16(a0, b3, acc[3]);
    acc[4] = WMMA_BF16(a1, b0, acc[4]);
    acc[5] = WMMA_BF16(a1, b1, acc[5]);
    acc[6] = WMMA_BF16(a1, b2, acc[6]);
    acc[7] = WMMA_BF16(a1, b3, acc[7]);

    a0 = a0n; a1 = a1n; b0 = b0n; b1 = b1n; b2 = b2n; b3 = b3n;
  }
}

// ---------------- GEMM kernels (block = 8 waves, 256x64 tile) ----------------

// A2 = 2*(g @ g) - I  (cheb recurrence, K=3), bf16 in/out, 512x512
__global__ __launch_bounds__(256) void gemm_cheb(
    const __bf16* __restrict__ A, const __bf16* __restrict__ Bt,
    __bf16* __restrict__ out)
{
  int wave = threadIdx.x >> 5, lane = threadIdx.x & 31;
  int row0 = blockIdx.x * 256 + wave * 32;
  int col0 = blockIdx.y * 64;
  v8f acc[8] = {};
  wmma_32x64(A, NP, Bt, NP, row0, col0, NP, lane, acc);
  int c = lane & 15, rb = (lane >> 4) * 8;
#pragma unroll
  for (int h = 0; h < 2; ++h)
#pragma unroll
    for (int t = 0; t < 4; ++t)
#pragma unroll
      for (int j = 0; j < 8; ++j) {
        int row = row0 + h * 16 + rb + j, col = col0 + t * 16 + c;
        float v = 2.f * acc[h * 4 + t][j];
        if (row == col && row < NN) v -= 1.f;
        out[(size_t)row * NP + col] = (__bf16)v;
      }
}

// Graph mix: Y_k = A_k @ X  for k in {1,2,4,5}; scatter into packed feature
// matrix XC2[(n*B+b)][k*Cin + c]  (rows = 32000, ld = ldx2)
__global__ __launch_bounds__(256) void gemm_mix(
    const __bf16* __restrict__ As, const __bf16* __restrict__ XtT,
    __bf16* __restrict__ XC2, int Cin, int ldx2)
{
  int kk = ((blockIdx.z & 1) + 1) + (blockIdx.z >> 1) * 3;  // 0,1,2,3 -> 1,2,4,5
  const __bf16* A = As + (size_t)kk * SMAT;
  int wave = threadIdx.x >> 5, lane = threadIdx.x & 31;
  int row0 = blockIdx.x * 256 + wave * 32;
  int col0 = blockIdx.y * 64;
  v8f acc[8] = {};
  wmma_32x64(A, NP, XtT, NP, row0, col0, NP, lane, acc);
  int c = lane & 15, rb = (lane >> 4) * 8;
#pragma unroll
  for (int h = 0; h < 2; ++h)
#pragma unroll
    for (int t = 0; t < 4; ++t)
#pragma unroll
      for (int j = 0; j < 8; ++j) {
        int n = row0 + h * 16 + rb + j;
        if (n < NN) {
          int jj = col0 + t * 16 + c;       // jj = b*Cin + cc
          int b  = jj / Cin;
          int cc = jj - b * Cin;
          XC2[((size_t)n * BB + b) * ldx2 + kk * Cin + cc] = (__bf16)acc[h * 4 + t][j];
        }
      }
}

// Dense stage: out = act(XC2 @ W + bias), fp32 out. act: 1=sigmoid 2=tanh
__global__ __launch_bounds__(256) void gemm_dense(
    const __bf16* __restrict__ A, const __bf16* __restrict__ WT,
    const float* __restrict__ bias, float* __restrict__ out,
    int K, int N, int act)
{
  int wave = threadIdx.x >> 5, lane = threadIdx.x & 31;
  int row0 = blockIdx.x * 256 + wave * 32;
  int col0 = blockIdx.y * 64;
  v8f acc[8] = {};
  wmma_32x64(A, K, WT, K, row0, col0, K, lane, acc);
  int c = lane & 15, rb = (lane >> 4) * 8;
#pragma unroll
  for (int h = 0; h < 2; ++h)
#pragma unroll
    for (int t = 0; t < 4; ++t)
#pragma unroll
      for (int j = 0; j < 8; ++j) {
        int row = row0 + h * 16 + rb + j, col = col0 + t * 16 + c;
        float v = acc[h * 4 + t][j] + bias[col];
        if (act == 1)      v = 1.f / (1.f + __expf(-v));
        else if (act == 2) v = tanhf(v);
        out[(size_t)row * N + col] = v;
      }
}

// ---------------- small / elementwise kernels ----------------

__global__ void zero_u32(unsigned* __restrict__ p, size_t n) {
  size_t i = (size_t)blockIdx.x * blockDim.x + threadIdx.x;
  if (i < n) p[i] = 0u;
}

// WT[o][f] = (f < K) ? W[f][o] : 0   (bf16, padded K -> Kp)
__global__ void wt_kernel(const float* __restrict__ W, __bf16* __restrict__ WT,
                          int K, int Nout, int Kp) {
  int tid = blockIdx.x * blockDim.x + threadIdx.x;
  if (tid >= Nout * Kp) return;
  int o = tid / Kp, f = tid - o * Kp;
  WT[tid] = (f < K) ? (__bf16)W[(size_t)f * Nout + o] : (__bf16)0.f;
}

// emb[n][d] = sum_j We[n][j] * Memory[j][d]
__global__ void emb_kernel(const float* __restrict__ We,
                           const float* __restrict__ Memory,
                           float* __restrict__ emb) {
  int tid = blockIdx.x * blockDim.x + threadIdx.x;
  if (tid >= NN * MEMD) return;
  int n = tid / MEMD, d = tid - n * MEMD;
  float s = 0.f;
  for (int j = 0; j < MEMN; ++j) s += We[n * MEMN + j] * Memory[j * MEMD + d];
  emb[tid] = s;
}

// g[n][m] = relu(dot(ea[n], eb[m]))
__global__ void score_kernel(const float* __restrict__ ea,
                             const float* __restrict__ eb,
                             float* __restrict__ g) {
  int tid = blockIdx.x * blockDim.x + threadIdx.x;
  if (tid >= NN * NN) return;
  int n = tid / NN, m = tid - n * NN;
  float s = 0.f;
  for (int d = 0; d < MEMD; ++d) s += ea[n * MEMD + d] * eb[m * MEMD + d];
  g[tid] = fmaxf(s, 0.f);
}

// row-wise softmax over 500 cols, in place
__global__ void softmax_rows(float* __restrict__ g) {
  int n = blockIdx.x, tid = threadIdx.x;
  __shared__ float red[256];
  float m = -1e30f;
  for (int j = tid; j < NN; j += 256) m = fmaxf(m, g[(size_t)n * NN + j]);
  red[tid] = m; __syncthreads();
  for (int s = 128; s > 0; s >>= 1) { if (tid < s) red[tid] = fmaxf(red[tid], red[tid + s]); __syncthreads(); }
  float mx = red[0]; __syncthreads();
  float sum = 0.f;
  for (int j = tid; j < NN; j += 256) {
    float e = __expf(g[(size_t)n * NN + j] - mx);
    g[(size_t)n * NN + j] = e; sum += e;
  }
  red[tid] = sum; __syncthreads();
  for (int s = 128; s > 0; s >>= 1) { if (tid < s) red[tid] += red[tid + s]; __syncthreads(); }
  float inv = 1.f / red[0];
  for (int j = tid; j < NN; j += 256) g[(size_t)n * NN + j] *= inv;
}

// As[base+0] = I, As[base+1] = g (bf16, zero-padded to 512), gT = g^T (bf16)
__global__ void prep_g(const float* __restrict__ g, __bf16* __restrict__ AsBase,
                       __bf16* __restrict__ gT) {
  int tid = blockIdx.x * blockDim.x + threadIdx.x;
  if (tid >= (int)SMAT) return;
  int n = tid >> 9, m = tid & (NP - 1);
  float v = (n < NN && m < NN) ? g[(size_t)n * NN + m] : 0.f;
  AsBase[tid]        = (__bf16)((n == m && n < NN) ? 1.f : 0.f);
  AsBase[SMAT + tid] = (__bf16)v;
  gT[(size_t)m * NP + n] = (__bf16)v;
}

// Build XtT[j][m] (transposed mix operand, bf16, m zero-padded to 512) from
// concat(x_or_go, (z*)state); also scatter identity cheb blocks k=0,3 into XC2.
__global__ void pack_kernel(const float* __restrict__ xin, int xin_is_go, int t,
                            const float* __restrict__ state,
                            const float* __restrict__ ZR, int use_z,
                            int H, int Cin, int J, int ldx2,
                            __bf16* __restrict__ XtT, __bf16* __restrict__ XC2)
{
  int tid = blockIdx.x * blockDim.x + threadIdx.x;
  if (tid >= J * NP) return;
  int j = tid >> 9;
  int m = tid & (NP - 1);
  int b = j / Cin;
  int c = j - b * Cin;
  float v = 0.f;
  if (m < NN) {
    if (c == 0) {
      v = xin_is_go ? xin[(size_t)b * NN + m]
                    : xin[((size_t)b * TT + t) * NN + m];
    } else {
      float s = state[((size_t)b * NN + m) * H + (c - 1)];
      if (use_z) s *= ZR[((size_t)m * BB + b) * (2 * H) + (c - 1)];
      v = s;
    }
  }
  __bf16 bv = (__bf16)v;
  XtT[(size_t)j * NP + m] = bv;
  if (m < NN) {
    size_t row = ((size_t)m * BB + b) * ldx2;
    XC2[row + c]           = bv;   // k = 0 (identity)
    XC2[row + 3 * Cin + c] = bv;   // k = 3 (identity)
  }
}

// state = r*state + (1-r)*hc  (r from ZR cols [H,2H), hc from HC)
__global__ void state_update(float* __restrict__ state,
                             const float* __restrict__ ZR,
                             const float* __restrict__ HC, int H) {
  int tid = blockIdx.x * blockDim.x + threadIdx.x;
  if (tid >= BB * NN * H) return;
  int h = tid % H;
  int n = (tid / H) % NN;
  int b = tid / (H * NN);
  size_t row = (size_t)n * BB + b;
  float r  = ZR[row * (2 * H) + H + h];
  float hc = HC[row * H + h];
  float s  = state[tid];
  state[tid] = r * s + (1.f - r) * hc;
}

// Memory attention: dec_state[b][n] = concat(h, softmax(hWq M^T) M)
__global__ void attention_kernel(const float* __restrict__ enc_state,
                                 const float* __restrict__ Wq,
                                 const float* __restrict__ Memory,
                                 float* __restrict__ dec_state) {
  int bn = blockIdx.x;             // b*500 + n
  int tid = threadIdx.x;           // 0..63
  __shared__ float h[64], q[64], att[MEMN];
  h[tid] = enc_state[(size_t)bn * HENC + tid];
  __syncthreads();
  float qv = 0.f;
  for (int c = 0; c < HENC; ++c) qv += h[c] * Wq[c * MEMD + tid];
  q[tid] = qv; __syncthreads();
  if (tid < MEMN) {
    float s = 0.f;
    for (int d = 0; d < MEMD; ++d) s += q[d] * Memory[tid * MEMD + d];
    att[tid] = s;
  }
  __syncthreads();
  if (tid == 0) {
    float mx = att[0];
    for (int j = 1; j < MEMN; ++j) mx = fmaxf(mx, att[j]);
    float sum = 0.f;
    for (int j = 0; j < MEMN; ++j) { att[j] = __expf(att[j] - mx); sum += att[j]; }
    float inv = 1.f / sum;
    for (int j = 0; j < MEMN; ++j) att[j] *= inv;
  }
  __syncthreads();
  float val = 0.f;
  for (int j = 0; j < MEMN; ++j) val += att[j] * Memory[j * MEMD + tid];
  size_t base = (size_t)bn * HDEC;
  dec_state[base + tid]      = h[tid];
  dec_state[base + 64 + tid] = val;
}

// go = st @ proj_W + proj_b ; also write output slice for step t
__global__ void proj_kernel(const float* __restrict__ dec_state,
                            const float* __restrict__ projW,
                            const float* __restrict__ projb,
                            float* __restrict__ go, float* __restrict__ out,
                            int t) {
  int tid = blockIdx.x * blockDim.x + threadIdx.x;
  if (tid >= BB * NN) return;
  int b = tid / NN, n = tid - b * NN;
  const float* s = dec_state + (size_t)tid * HDEC;
  float acc = projb[0];
  for (int h = 0; h < HDEC; ++h) acc += s[h] * projW[h];
  go[tid] = acc;
  out[((size_t)b * TT + t) * NN + n] = acc;
}

// ---------------- host orchestration ----------------

static inline dim3 g1d(size_t n, int blk) { return dim3((unsigned)((n + blk - 1) / blk)); }

extern "C" void kernel_launch(void* const* d_in, const int* in_sizes, int n_in,
                              void* d_out, int out_size, void* d_ws, size_t ws_size,
                              hipStream_t stream) {
  const float* x          = (const float*)d_in[0];
  const float* Memory     = (const float*)d_in[1];
  const float* Wq         = (const float*)d_in[2];
  const float* We1        = (const float*)d_in[3];
  const float* We2        = (const float*)d_in[4];
  const float* enc_gate_W = (const float*)d_in[5];
  const float* enc_gate_b = (const float*)d_in[6];
  const float* enc_up_W   = (const float*)d_in[7];
  const float* enc_up_b   = (const float*)d_in[8];
  const float* dec_gate_W = (const float*)d_in[9];
  const float* dec_gate_b = (const float*)d_in[10];
  const float* dec_up_W   = (const float*)d_in[11];
  const float* dec_up_b   = (const float*)d_in[12];
  const float* proj_W     = (const float*)d_in[13];
  const float* proj_b     = (const float*)d_in[14];
  float* out = (float*)d_out;

  // geometry
  const int CinE = 1 + HENC;           // 65
  const int CinD = 1 + HDEC;           // 129
  const int JE   = BB * CinE;          // 4160
  const int JD   = BB * CinD;          // 8256
  const int KfE  = 416;                // 6*65=390 -> pad 416 (mult of 32)
  const int KfD  = 800;                // 6*129=774 -> pad 800
  const size_t ROWS = (size_t)NN * BB; // 32000

  // workspace layout
  char* w = (char*)d_ws;
  size_t off = 0;
  auto alloc = [&](size_t bytes) -> void* {
    void* p = (void*)(w + off);
    off = (off + bytes + 255) & ~(size_t)255;
    return p;
  };
  __bf16* As   = (__bf16*)alloc(6 * SMAT * 2);
  __bf16* gT   = (__bf16*)alloc(SMAT * 2);
  float*  g1   = (float*) alloc((size_t)NN * NN * 4);
  float*  g2   = (float*) alloc((size_t)NN * NN * 4);
  float*  emb1 = (float*) alloc((size_t)NN * MEMD * 4);
  float*  emb2 = (float*) alloc((size_t)NN * MEMD * 4);
  __bf16* WTeg = (__bf16*)alloc((size_t)128 * KfE * 2);
  __bf16* WTeu = (__bf16*)alloc((size_t)64  * KfE * 2);
  __bf16* WTdg = (__bf16*)alloc((size_t)256 * KfD * 2);
  __bf16* WTdu = (__bf16*)alloc((size_t)128 * KfD * 2);
  __bf16* XtT  = (__bf16*)alloc((size_t)JD * NP * 2);
  __bf16* XC2  = (__bf16*)alloc(ROWS * KfD * 2);
  float*  ZR   = (float*) alloc(ROWS * 256 * 4);
  float*  HC   = (float*) alloc(ROWS * 128 * 4);
  float*  encS = (float*) alloc((size_t)BB * NN * HENC * 4);
  float*  decS = (float*) alloc((size_t)BB * NN * HDEC * 4);
  float*  go   = (float*) alloc((size_t)BB * NN * 4);
  (void)ws_size; (void)n_in; (void)in_sizes; (void)out_size;

  // ---- init / weight prep ----
  zero_u32<<<g1d((size_t)BB * NN * HENC, 256), 256, 0, stream>>>((unsigned*)encS, (size_t)BB * NN * HENC);
  zero_u32<<<g1d((size_t)BB * NN, 256), 256, 0, stream>>>((unsigned*)go, (size_t)BB * NN);
  wt_kernel<<<g1d((size_t)128 * KfE, 256), 256, 0, stream>>>(enc_gate_W, WTeg, 390, 128, KfE);
  wt_kernel<<<g1d((size_t)64  * KfE, 256), 256, 0, stream>>>(enc_up_W,   WTeu, 390, 64,  KfE);
  wt_kernel<<<g1d((size_t)256 * KfD, 256), 256, 0, stream>>>(dec_gate_W, WTdg, 774, 256, KfD);
  wt_kernel<<<g1d((size_t)128 * KfD, 256), 256, 0, stream>>>(dec_up_W,   WTdu, 774, 128, KfD);

  // ---- adaptive graph + cheb stack ----
  emb_kernel<<<g1d((size_t)NN * MEMD, 256), 256, 0, stream>>>(We1, Memory, emb1);
  emb_kernel<<<g1d((size_t)NN * MEMD, 256), 256, 0, stream>>>(We2, Memory, emb2);
  score_kernel<<<g1d((size_t)NN * NN, 256), 256, 0, stream>>>(emb1, emb2, g1);
  score_kernel<<<g1d((size_t)NN * NN, 256), 256, 0, stream>>>(emb2, emb1, g2);
  softmax_rows<<<NN, 256, 0, stream>>>(g1);
  softmax_rows<<<NN, 256, 0, stream>>>(g2);
  prep_g<<<g1d(SMAT, 256), 256, 0, stream>>>(g1, As, gT);
  gemm_cheb<<<dim3(NP / 256, NP / 64), 256, 0, stream>>>(As + SMAT, gT, As + 2 * SMAT);
  prep_g<<<g1d(SMAT, 256), 256, 0, stream>>>(g2, As + 3 * SMAT, gT);
  gemm_cheb<<<dim3(NP / 256, NP / 64), 256, 0, stream>>>(As + 4 * SMAT, gT, As + 5 * SMAT);

  // ---- encoder: 12 AGCRN steps (hidden 64) ----
  zero_u32<<<g1d(ROWS * KfD / 2, 256), 256, 0, stream>>>((unsigned*)XC2, ROWS * KfD / 2);
  dim3 mixE(NP / 256, JE / 64, 4);
  dim3 dgridG(ROWS / 256, 128 / 64);   // gate out 128
  dim3 dgridU(ROWS / 256, 64 / 64);    // update out 64
  for (int t = 0; t < TT; ++t) {
    pack_kernel<<<g1d((size_t)JE * NP, 256), 256, 0, stream>>>(
        x, 0, t, encS, ZR, 0, HENC, CinE, JE, KfE, XtT, XC2);
    gemm_mix<<<mixE, 256, 0, stream>>>(As, XtT, XC2, CinE, KfE);
    gemm_dense<<<dgridG, 256, 0, stream>>>(XC2, WTeg, enc_gate_b, ZR, KfE, 128, 1);
    pack_kernel<<<g1d((size_t)JE * NP, 256), 256, 0, stream>>>(
        x, 0, t, encS, ZR, 1, HENC, CinE, JE, KfE, XtT, XC2);
    gemm_mix<<<mixE, 256, 0, stream>>>(As, XtT, XC2, CinE, KfE);
    gemm_dense<<<dgridU, 256, 0, stream>>>(XC2, WTeu, enc_up_b, HC, KfE, 64, 2);
    state_update<<<g1d((size_t)BB * NN * HENC, 256), 256, 0, stream>>>(encS, ZR, HC, HENC);
  }

  // ---- memory attention -> decoder initial state (hidden 128) ----
  attention_kernel<<<BB * NN, 64, 0, stream>>>(encS, Wq, Memory, decS);

  // ---- decoder: 12 AGCRN steps (hidden 128) + projection ----
  zero_u32<<<g1d(ROWS * KfD / 2, 256), 256, 0, stream>>>((unsigned*)XC2, ROWS * KfD / 2);
  dim3 mixD(NP / 256, JD / 64, 4);
  dim3 dgridDG(ROWS / 256, 256 / 64);  // gate out 256
  dim3 dgridDU(ROWS / 256, 128 / 64);  // update out 128
  for (int t = 0; t < TT; ++t) {
    pack_kernel<<<g1d((size_t)JD * NP, 256), 256, 0, stream>>>(
        go, 1, 0, decS, ZR, 0, HDEC, CinD, JD, KfD, XtT, XC2);
    gemm_mix<<<mixD, 256, 0, stream>>>(As, XtT, XC2, CinD, KfD);
    gemm_dense<<<dgridDG, 256, 0, stream>>>(XC2, WTdg, dec_gate_b, ZR, KfD, 256, 1);
    pack_kernel<<<g1d((size_t)JD * NP, 256), 256, 0, stream>>>(
        go, 1, 0, decS, ZR, 1, HDEC, CinD, JD, KfD, XtT, XC2);
    gemm_mix<<<mixD, 256, 0, stream>>>(As, XtT, XC2, CinD, KfD);
    gemm_dense<<<dgridDU, 256, 0, stream>>>(XC2, WTdu, dec_up_b, HC, KfD, 128, 2);
    state_update<<<g1d((size_t)BB * NN * HDEC, 256), 256, 0, stream>>>(decS, ZR, HC, HDEC);
    proj_kernel<<<g1d((size_t)BB * NN, 256), 256, 0, stream>>>(decS, proj_W, proj_b, go, out, t);
  }
}
